// ContrastiveLoss_44298292691534
// MI455X (gfx1250) — matrix operand order repaired and checked by reading, not compiled, not run on hardware
//
#include <hip/hip_runtime.h>
#include <math.h>

// ---------------------------------------------------------------------------
// ContrastiveLoss on MI455X (gfx1250, wave32, WMMA)
//   pos = mean_i ||o2_i - o1_i||^2
//   dist[i][j] = sqrt(max(|o1_i|^2 + |o2_j|^2 - 2 o1_i.o2_j, 0))
//   neg = mean_i hinge(MARGIN - rank_rn(dist[i,:]))   (self-reject rule)
// GEMM on v_wmma_f32_16x16x32_bf16, fp32 accumulate. B pre-converted to bf16
// in workspace. Each wave: 4 M-tiles x 2 N-tiles = 8 WMMAs per K-step sharing
// 4 A fragments (1 ds_load_b128/WMMA) and 2 B fragments (0.5 vmem/WMMA).
// ---------------------------------------------------------------------------

typedef __attribute__((ext_vector_type(16))) __bf16 v16bf;
typedef __attribute__((ext_vector_type(8)))  float  v8f;

#define N_ROWS   8192
#define DIM      1024
#define QUANT    30
#define MARGIN_F 2.0f
#define TOPK     32            // keep 32 smallest; first 30 == reference top-30
#define MROWS    64            // rows (M) per workgroup (4 WMMA M-tiles)
#define MTILES   (MROWS / 16)
#define NTILES   2             // 16-col tiles per wave sharing A fragments
#define WAVES    8
#define NTHREADS 256
#define CTILE    (WAVES * 16 * NTILES)  // 256 columns per sweep iteration
#define A_STRIDE (DIM + 8)     // bf16 elems; pad so 16 rows hit distinct banks
#define NSPLIT   2             // column splits (occupancy vs L2 traffic)

// ---------------------------------------------------------------------------
__global__ void cl_init_kernel(float* __restrict__ accum) {
  accum[0] = 0.0f;  // pos-loss sum
  accum[1] = 0.0f;  // neg-loss sum
}

// ---------------------------------------------------------------------------
// Per-row norms + positive-loss sum. One block per row.
__global__ __launch_bounds__(256) void cl_rownorm_kernel(
    const float* __restrict__ o1, const float* __restrict__ o2,
    float* __restrict__ a2, float* __restrict__ b2,
    float* __restrict__ accum) {
  const int row = blockIdx.x;
  const float* r1 = o1 + (size_t)row * DIM;
  const float* r2 = o2 + (size_t)row * DIM;
  float pa = 0.f, pb = 0.f, pp = 0.f;
  for (int c = threadIdx.x; c < DIM; c += blockDim.x) {
    float x = r1[c], y = r2[c], d = y - x;
    pa += x * x; pb += y * y; pp += d * d;
  }
  __shared__ float sa[256], sb[256], sp[256];
  sa[threadIdx.x] = pa; sb[threadIdx.x] = pb; sp[threadIdx.x] = pp;
  __syncthreads();
  for (int s = 128; s > 0; s >>= 1) {
    if ((int)threadIdx.x < s) {
      sa[threadIdx.x] += sa[threadIdx.x + s];
      sb[threadIdx.x] += sb[threadIdx.x + s];
      sp[threadIdx.x] += sp[threadIdx.x + s];
    }
    __syncthreads();
  }
  if (threadIdx.x == 0) {
    a2[row] = sa[0];
    b2[row] = sb[0];
    atomicAdd(&accum[0], sp[0]);
  }
}

// ---------------------------------------------------------------------------
// fp32 -> bf16 conversion of o2 (8 elems / thread, 16B stores).
__global__ __launch_bounds__(256) void cl_cvt_bf16_kernel(
    const float* __restrict__ in, __bf16* __restrict__ out) {
  const size_t i = ((size_t)blockIdx.x * blockDim.x + threadIdx.x) * 8;
  const float4 a = ((const float4*)(in + i))[0];
  const float4 b = ((const float4*)(in + i))[1];
  __bf16 t[8] = {(__bf16)a.x, (__bf16)a.y, (__bf16)a.z, (__bf16)a.w,
                 (__bf16)b.x, (__bf16)b.y, (__bf16)b.z, (__bf16)b.w};
  *((uint4*)(out + i)) = *(uint4*)t;
}

// ---------------------------------------------------------------------------
// Helper: build a bf16 B fragment from fp32 memory (fallback path).
__device__ __forceinline__ v16bf cl_b_frag_f32(const float* bp_) {
  const float4* bp = (const float4*)bp_;
  const float4 q0 = bp[0], q1 = bp[1], q2 = bp[2], q3 = bp[3];
  v16bf f;
  f[0]  = (__bf16)q0.x; f[1]  = (__bf16)q0.y;
  f[2]  = (__bf16)q0.z; f[3]  = (__bf16)q0.w;
  f[4]  = (__bf16)q1.x; f[5]  = (__bf16)q1.y;
  f[6]  = (__bf16)q1.z; f[7]  = (__bf16)q1.w;
  f[8]  = (__bf16)q2.x; f[9]  = (__bf16)q2.y;
  f[10] = (__bf16)q2.z; f[11] = (__bf16)q2.w;
  f[12] = (__bf16)q3.x; f[13] = (__bf16)q3.y;
  f[14] = (__bf16)q3.z; f[15] = (__bf16)q3.w;
  return f;
}

// ---------------------------------------------------------------------------
// Fused GEMM (bf16 WMMA) + distance + per-row 32-smallest selection.
// WRITE_PARTIAL: emit sorted partial lists to global (column-split mode);
// otherwise do rank-rn pick + hinge in-kernel (single-split fallback).
// B_BF16: B fragments loaded directly as bf16 from workspace.
template <bool WRITE_PARTIAL, bool B_BF16>
__global__ __launch_bounds__(NTHREADS) void cl_gram_topk_kernel(
    const float* __restrict__ o1, const float* __restrict__ o2f,
    const __bf16* __restrict__ o2h, const int* __restrict__ rn,
    const float* __restrict__ a2, const float* __restrict__ b2,
    float* __restrict__ accum,
    float* __restrict__ ptv, int* __restrict__ pti) {
  __shared__ __bf16 Abf[MROWS * A_STRIDE];      // 129 KB bf16 A tile
  __shared__ float  distS[MROWS][CTILE + 4];    // 66 KB distance staging
  __shared__ float  topv[MROWS][TOPK];
  __shared__ int    topi[MROWS][TOPK];
  __shared__ float  a2s[MROWS];

  const int tid   = threadIdx.x;
  const int row0  = blockIdx.x * MROWS;
  const int nspl  = gridDim.y;
  const int ncols = N_ROWS / nspl;
  const int cbase = blockIdx.y * ncols;

  // Stage A rows (fp32 -> bf16) into LDS, 8 elems / thread / step.
  for (int base = tid * 8; base < MROWS * DIM; base += NTHREADS * 8) {
    const int r = base >> 10;          // DIM == 1024
    const int c = base & (DIM - 1);
    const float* src = o1 + (size_t)(row0 + r) * DIM + c;
    const float4 x = ((const float4*)src)[0];
    const float4 y = ((const float4*)src)[1];
    __bf16 t[8] = {(__bf16)x.x, (__bf16)x.y, (__bf16)x.z, (__bf16)x.w,
                   (__bf16)y.x, (__bf16)y.y, (__bf16)y.z, (__bf16)y.w};
    *((uint4*)&Abf[r * A_STRIDE + c]) = *(uint4*)t;
  }
  for (int idx = tid; idx < MROWS * TOPK; idx += NTHREADS) {
    topv[idx >> 5][idx & (TOPK - 1)] = 3.4e38f;
    topi[idx >> 5][idx & (TOPK - 1)] = -1;
  }
  if (tid < MROWS) a2s[tid] = a2[row0 + tid];
  __syncthreads();

  const int wave  = tid >> 5;
  const int lane  = tid & 31;
  const int lhalf = lane >> 4;        // 0 | 1
  const int l15   = lane & 15;
  const int koffA = lhalf << 4;       // A frag: lanes 16-31 hold K 16..31
  const int koffB = lhalf << 4;       // B frag: lanes 16-31 hold K 16..31

  for (int c0 = 0; c0 < ncols; c0 += CTILE) {
    // Two 16-wide column tiles per wave, adjacent, sharing A fragments.
    const int col0 = cbase + c0 + wave * (16 * NTILES) + l15;
    const int col1 = col0 + 16;
    const float*  bf0 = o2f ? o2f + (size_t)col0 * DIM + koffB : nullptr;
    const float*  bf1 = o2f ? o2f + (size_t)col1 * DIM + koffB : nullptr;
    const __bf16* bh0 = o2h ? o2h + (size_t)col0 * DIM + koffB : nullptr;
    const __bf16* bh1 = o2h ? o2h + (size_t)col1 * DIM + koffB : nullptr;
    if (c0 + CTILE < ncols) {                        // warm next tile's B rows
      if (B_BF16) {
        __builtin_prefetch((const void*)(bh0 + (size_t)CTILE * DIM), 0, 0);
        __builtin_prefetch((const void*)(bh1 + (size_t)CTILE * DIM), 0, 0);
      } else {
        __builtin_prefetch((const void*)(bf0 + (size_t)CTILE * DIM), 0, 0);
        __builtin_prefetch((const void*)(bf1 + (size_t)CTILE * DIM), 0, 0);
      }
    }

    v8f acc[NTILES][MTILES];
#pragma unroll
    for (int nt = 0; nt < NTILES; ++nt)
#pragma unroll
      for (int mt = 0; mt < MTILES; ++mt) acc[nt][mt] = (v8f){};

#pragma unroll 2
    for (int k = 0; k < DIM; k += 32) {
      // --- 2 B fragments (16 K-contiguous values per lane each) ---
      v16bf bfrag0, bfrag1;
      if (B_BF16) {
        bfrag0 = *(const v16bf*)(bh0 + k);           // 2x global_load_b128
        bfrag1 = *(const v16bf*)(bh1 + k);
      } else {
        bfrag0 = cl_b_frag_f32(bf0 + k);
        bfrag1 = cl_b_frag_f32(bf1 + k);
      }
      // --- 4 A fragments from LDS; each feeds 2 WMMAs (one per N-tile) ---
#pragma unroll
      for (int mt = 0; mt < MTILES; ++mt) {
        const __bf16* ap = Abf + (mt * 16 + l15) * A_STRIDE + k + koffA;
        v16bf af;
#pragma unroll
        for (int j = 0; j < 8; ++j) {
          af[j]     = ap[j];
          af[8 + j] = ap[8 + j];
        }
        acc[0][mt] = __builtin_amdgcn_wmma_f32_16x16x32_bf16(
            false, af, false, bfrag0, (short)0, acc[0][mt], false, false);
        acc[1][mt] = __builtin_amdgcn_wmma_f32_16x16x32_bf16(
            false, af, false, bfrag1, (short)0, acc[1][mt], false, false);
      }
    }

    // --- distances for this 64x32 tile pair ---
    const float bb0 = b2[col0];
    const float bb1 = b2[col1];
    const int   ct0 = wave * (16 * NTILES) + l15;
    const int   ct1 = ct0 + 16;
#pragma unroll
    for (int mt = 0; mt < MTILES; ++mt) {
#pragma unroll
      for (int r = 0; r < 8; ++r) {
        const int rowA = mt * 16 + (lhalf << 3) + r;   // C layout: M=r+8*lhalf
        const float sq0 = a2s[rowA] + bb0 - 2.0f * acc[0][mt][r];
        const float sq1 = a2s[rowA] + bb1 - 2.0f * acc[1][mt][r];
        distS[rowA][ct0] = sqrtf(fmaxf(sq0, 0.0f));
        distS[rowA][ct1] = sqrtf(fmaxf(sq1, 0.0f));
      }
    }
    __syncthreads();

    // --- per-row 32-smallest maintenance (one thread per row, b128 scan) ---
    if (tid < MROWS) {
      float* tv = topv[tid];
      int*   ti = topi[tid];
      float worst = tv[TOPK - 1];
      const float4* dp = (const float4*)distS[tid];
#pragma unroll 4
      for (int j4 = 0; j4 < CTILE / 4; ++j4) {
        const float4 q = dp[j4];
        const float vals[4] = {q.x, q.y, q.z, q.w};
#pragma unroll
        for (int u = 0; u < 4; ++u) {
          const float v = vals[u];
          if (v < worst) {
            int p = TOPK - 1;
            while (p > 0 && tv[p - 1] > v) {
              tv[p] = tv[p - 1]; ti[p] = ti[p - 1]; --p;
            }
            tv[p] = v; ti[p] = cbase + c0 + j4 * 4 + u;
            worst = tv[TOPK - 1];
          }
        }
      }
    }
    __syncthreads();
  }

  if (WRITE_PARTIAL) {
    // Emit sorted partial lists: layout [row][split][TOPK].
    for (int idx = tid; idx < MROWS * TOPK; idx += NTHREADS) {
      const int r = idx >> 5, t = idx & (TOPK - 1);
      const size_t o = ((size_t)(row0 + r) * nspl + blockIdx.y) * TOPK + t;
      ptv[o] = topv[r][t];
      pti[o] = topi[r][t];
    }
  } else {
    // Single-split: rank-rn pick with self-match rejection, hinge.
    if (tid < MROWS) {
      const int grow = row0 + tid;
      const int r    = rn[grow];
      const int sel  = topi[tid][r];
      const int rs   = (sel == grow) ? ((r + 1) % QUANT) : r;
      atomicAdd(&accum[1], fmaxf(MARGIN_F - topv[tid][rs], 0.0f));
    }
  }
}

// ---------------------------------------------------------------------------
// 2-way merge of sorted partial lists -> rank-rn pick -> hinge accumulate.
__global__ __launch_bounds__(256) void cl_merge_kernel(
    const float* __restrict__ ptv, const int* __restrict__ pti,
    const int* __restrict__ rn, float* __restrict__ accum) {
  const int row = blockIdx.x * blockDim.x + threadIdx.x;
  const float* v0 = ptv + ((size_t)row * NSPLIT + 0) * TOPK;
  const float* v1 = ptv + ((size_t)row * NSPLIT + 1) * TOPK;
  const int*   i0 = pti + ((size_t)row * NSPLIT + 0) * TOPK;
  const int*   i1 = pti + ((size_t)row * NSPLIT + 1) * TOPK;

  const int r    = rn[row];
  const int ralt = (r + 1) % QUANT;
  float v_r = 0.f, v_alt = 0.f;
  int   i_r = -1;
  int a = 0, b = 0;
  for (int t = 0; t < QUANT; ++t) {
    float v; int ix;
    if (v0[a] <= v1[b]) { v = v0[a]; ix = i0[a]; ++a; }
    else                { v = v1[b]; ix = i1[b]; ++b; }
    if (t == r)    { v_r = v; i_r = ix; }
    if (t == ralt) { v_alt = v; }
  }
  const float v = (i_r == row) ? v_alt : v_r;
  atomicAdd(&accum[1], fmaxf(MARGIN_F - v, 0.0f));
}

// ---------------------------------------------------------------------------
__global__ void cl_finalize_kernel(const float* __restrict__ accum,
                                   float* __restrict__ out) {
  out[0] = accum[0] * (1.0f / (float)N_ROWS) +
           accum[1] * (1.0f / (float)N_ROWS);
}

// ---------------------------------------------------------------------------
extern "C" void kernel_launch(void* const* d_in, const int* in_sizes, int n_in,
                              void* d_out, int out_size, void* d_ws,
                              size_t ws_size, hipStream_t stream) {
  const float* o1 = (const float*)d_in[0];
  const float* o2 = (const float*)d_in[1];
  const int*   rn = (const int*)d_in[2];
  float* out = (float*)d_out;

  // ws layout: accum[2] | a2[N] | b2[N] | ptv | pti | o2 bf16 copy
  char*  ws    = (char*)d_ws;
  float* accum = (float*)ws;
  float* a2    = accum + 2;
  float* b2    = a2 + N_ROWS;
  size_t off   = (2 + 2 * (size_t)N_ROWS) * sizeof(float);
  off = (off + 255) & ~(size_t)255;
  float* ptv = (float*)(ws + off);
  off += (size_t)N_ROWS * NSPLIT * TOPK * sizeof(float);
  int* pti = (int*)(ws + off);
  off += (size_t)N_ROWS * NSPLIT * TOPK * sizeof(int);
  off = (off + 255) & ~(size_t)255;
  __bf16* o2h = (__bf16*)(ws + off);
  const size_t need_fast = off + (size_t)N_ROWS * DIM * sizeof(__bf16);

  cl_init_kernel<<<1, 1, 0, stream>>>(accum);
  cl_rownorm_kernel<<<N_ROWS, 256, 0, stream>>>(o1, o2, a2, b2, accum);

  if (ws_size >= need_fast) {
    // Fast path: bf16 B in ws, 2-way column split, merge pass.
    cl_cvt_bf16_kernel<<<(N_ROWS * DIM) / (256 * 8), 256, 0, stream>>>(o2, o2h);
    dim3 grid(N_ROWS / MROWS, NSPLIT);
    cl_gram_topk_kernel<true, true><<<grid, NTHREADS, 0, stream>>>(
        o1, nullptr, o2h, rn, a2, b2, accum, ptv, pti);
    cl_merge_kernel<<<N_ROWS / 256, 256, 0, stream>>>(ptv, pti, rn, accum);
  } else {
    // Fallback: fp32 B on the fly, single split, in-kernel pick (64 KB ws).
    dim3 grid(N_ROWS / MROWS, 1);
    cl_gram_topk_kernel<false, false><<<grid, NTHREADS, 0, stream>>>(
        o1, o2, nullptr, rn, a2, b2, accum, nullptr, nullptr);
  }
  cl_finalize_kernel<<<1, 1, 0, stream>>>(accum, out);
}
